// Model_10024453669634
// MI455X (gfx1250) — matrix operand order repaired and checked by reading, not compiled, not run on hardware
//
#include <hip/hip_runtime.h>
#include <cstdint>

// ---------------------------------------------------------------------------
// Atomistic NN potential: per-species 4-layer MLP + scatter-add.
//   Pre-pass: W1/W2/W3 f32 [K][N] -> bf16 [N][K] (transposed, WMMA-friendly)
//   L1: [4,8192,7584] x W1t  (bf16 WMMA, f32 accum, SiLU); features cvt in-kernel,
//       streamed non-temporal, double-buffered LDS pipeline.
//   L2/L3: [4,8192,256] x W2t/W3t, fully async-staged bf16, double-buffered.
//   L4: dot(h,W4)+b4 per atom, atomicAdd into per-structure energy.
// ---------------------------------------------------------------------------

typedef __attribute__((ext_vector_type(16))) __bf16 v16bf;
typedef __attribute__((ext_vector_type(8)))  __bf16 v8bf;
typedef __attribute__((ext_vector_type(4)))  __bf16 v4bf;
typedef __attribute__((ext_vector_type(8)))  float  v8f;
typedef __attribute__((ext_vector_type(4)))  float  v4f;

static __device__ __forceinline__ v8f wmma_bf16(v16bf a, v16bf b, v8f c) {
  return __builtin_amdgcn_wmma_f32_16x16x32_bf16(
      /*neg_a=*/false, a, /*neg_b=*/false, b,
      /*c_mod=*/(short)0, c, /*reuse_a=*/false, /*reuse_b=*/false);
}

// Async global -> LDS copy (CDNA5, ASYNCcnt-tracked). LDS address is the low
// 32 bits of the generic pointer (LDS aperture keeps the offset in [31:0]).
static __device__ __forceinline__ void async_load_b128(const void* gptr,
                                                       const void* lptr) {
  unsigned lds = (unsigned)(uintptr_t)lptr;
  asm volatile("global_load_async_to_lds_b128 %0, %1, off"
               :: "v"(lds), "v"(gptr) : "memory");
}
static __device__ __forceinline__ void wait_async0() {
  asm volatile("s_wait_asynccnt 0x0" ::: "memory");
}

// ---------------------------------------------------------------------------
// Weight pre-pass: W[s][K][N] f32 -> Wt[s][N][K] bf16 (32x32 LDS transpose).
// ---------------------------------------------------------------------------
__global__ __launch_bounds__(256)
void transpose_w_bf16(const float* __restrict__ in, __bf16* __restrict__ out,
                      int K, int N) {
  __shared__ __bf16 t[32][33];
  const int k0 = blockIdx.x * 32, n0 = blockIdx.y * 32, s = blockIdx.z;
  const float* gi = in + (size_t)s * K * N;
  __bf16* go = out + (size_t)s * N * K;
  const int x = threadIdx.x;  // 0..31
#pragma unroll
  for (int y = threadIdx.y; y < 32; y += 8)
    t[y][x] = (__bf16)gi[(size_t)(k0 + y) * N + n0 + x];  // coalesced in n
  __syncthreads();
#pragma unroll
  for (int y = threadIdx.y; y < 32; y += 8)
    go[(size_t)(n0 + y) * K + k0 + x] = t[x][y];          // coalesced in k
}

// ---------------------------------------------------------------------------
// Grouped GEMM + bias + SiLU: Out[s] = silu(A[s](MxK) * Wt[s](NxK)^T + b[s])
//   BM=128, BN=256 (full hidden width -> A streamed exactly once).
//   512 threads = 16 waves; each wave computes a 32x64 subtile (2x4 WMMA tiles).
//   Double-buffered LDS: next tile's async copies (and register-held f32
//   feature loads) are issued before computing the current tile.
// ---------------------------------------------------------------------------
template <typename AT, int BK>
__global__ __launch_bounds__(512)
void gemm_bias_silu(const AT* __restrict__ A, const __bf16* __restrict__ Wt,
                    const float* __restrict__ bias, __bf16* __restrict__ Out,
                    int M, int K, int N) {
  constexpr int BM = 128, BN = 256, PAD = 8, T = 512;
  constexpr bool A_F32 = (sizeof(AT) == 4);
  constexpr int PASS_A = (BM * BK) / (T * 4);   // f32 path: float4 per pass
  constexpr int CPR    = BK / 8;                // 16B chunks per LDS row
  constexpr int CH_A   = (BM * BK) / (T * 8);   // bf16 path: b128 chunks
  constexpr int CH_B   = (BN * BK) / (T * 8);

  __shared__ __bf16 sA[2][BM][BK + PAD];
  __shared__ __bf16 sB[2][BN][BK + PAD];   // Wt rows: [n][k], contiguous in k

  const int s    = blockIdx.y;
  const int m0   = blockIdx.x * BM;
  const int tid  = threadIdx.x;
  const int lane = tid & 31;
  const int wave = tid >> 5;
  const int wm   = wave & 3;    // 32-row quarter of BM
  const int wn   = wave >> 2;   // 64-col quarter of BN
  const int l    = lane & 15;
  const int hi   = lane >> 4;   // lane half selects K sub-block (k = 16*hi + s, contiguous order)

  const AT*     gA = A    + (size_t)s * M * K;
  const __bf16* gB = Wt   + (size_t)s * N * K;
  const float*  gb = bias + (size_t)s * N;
  __bf16*       gO = Out  + (size_t)s * M * N;

  v8f acc[2][4];
#pragma unroll
  for (int f = 0; f < 2; ++f)
#pragma unroll
    for (int g = 0; g < 4; ++g) acc[f][g] = (v8f)(0.0f);

  v4f regsA[A_F32 ? PASS_A : 1];

  auto issueB = [&](int buf, int kb) {
#pragma unroll
    for (int p = 0; p < CH_B; ++p) {
      int id  = tid + p * T;
      int row = id / CPR, c = (id % CPR) * 8;
      async_load_b128(&gB[(size_t)row * K + kb + c], &sB[buf][row][c]);
    }
  };
  auto issueA = [&](int buf, int kb) {  // bf16-activation path only
#pragma unroll
    for (int p = 0; p < CH_A; ++p) {
      int id  = tid + p * T;
      int row = id / CPR, c = (id % CPR) * 8;
      async_load_b128(&gA[(size_t)(m0 + row) * K + kb + c], &sA[buf][row][c]);
    }
  };
  auto loadAregs = [&](int kb) {  // f32 feature path: NT stream into registers
#pragma unroll
    for (int p = 0; p < PASS_A; ++p) {
      int idx = (tid + p * T) * 4;
      int row = idx / BK, col = idx % BK;
      const v4f* gp = (const v4f*)&gA[(size_t)(m0 + row) * K + kb + col];
#if __has_builtin(__builtin_nontemporal_load)
      regsA[p] = __builtin_nontemporal_load(gp);
#else
      regsA[p] = *gp;
#endif
    }
  };
  auto storeAregs = [&](int buf) {  // convert + store into LDS
#pragma unroll
    for (int p = 0; p < PASS_A; ++p) {
      int idx = (tid + p * T) * 4;
      int row = idx / BK, col = idx % BK;
      v4f v = regsA[p];
      v4bf b;
      b[0] = (__bf16)v[0]; b[1] = (__bf16)v[1];
      b[2] = (__bf16)v[2]; b[3] = (__bf16)v[3];
      *(v4bf*)&sA[buf][row][col] = b;
    }
  };
  auto compute = [&](int buf) {
#pragma unroll
    for (int kk = 0; kk < BK; kk += 32) {
      v16bf bfrag[4];
#pragma unroll
      for (int g = 0; g < 4; ++g) {
        const int c = wn * 64 + g * 16 + l;
        v8bf x0 = *(const v8bf*)&sB[buf][c][kk + hi * 16];
        v8bf x1 = *(const v8bf*)&sB[buf][c][kk + hi * 16 + 8];
        bfrag[g] = __builtin_shufflevector(
            x0, x1, 0, 1, 2, 3, 4, 5, 6, 7, 8, 9, 10, 11, 12, 13, 14, 15);
      }
#pragma unroll
      for (int f = 0; f < 2; ++f) {
        const int r = wm * 32 + f * 16 + l;
        v8bf a0 = *(const v8bf*)&sA[buf][r][kk + hi * 16];
        v8bf a1 = *(const v8bf*)&sA[buf][r][kk + hi * 16 + 8];
        v16bf af = __builtin_shufflevector(
            a0, a1, 0, 1, 2, 3, 4, 5, 6, 7, 8, 9, 10, 11, 12, 13, 14, 15);
#pragma unroll
        for (int g = 0; g < 4; ++g)
          acc[f][g] = wmma_bf16(af, bfrag[g], acc[f][g]);
      }
    }
  };

  // ---- prologue: fill buffer 0 ----
  issueB(0, 0);
  if constexpr (A_F32) {
    loadAregs(0);
    storeAregs(0);
  } else {
    issueA(0, 0);
  }
  wait_async0();
  __syncthreads();

  // ---- pipelined main loop: one barrier per K-tile ----
  int cur = 0;
  for (int kb = 0; kb < K; kb += BK) {
    const bool has_next = (kb + BK) < K;
    if (has_next) {
      issueB(cur ^ 1, kb + BK);               // async into free buffer
      if constexpr (A_F32) loadAregs(kb + BK);  // NT global loads in flight
      else issueA(cur ^ 1, kb + BK);
    }
    compute(cur);                             // WMMAs overlap the loads above
    if (has_next) {
      if constexpr (A_F32) storeAregs(cur ^ 1);
      wait_async0();                          // next buffer's copies landed
    }
    __syncthreads();                          // cur freed, cur^1 published
    cur ^= 1;
  }

  // ---- epilogue: bias + SiLU, store bf16 ----
  // C/D layout: VGPR r -> M = r + hi*8 ; N-col = lane&15
#pragma unroll
  for (int g = 0; g < 4; ++g) {
    const int col = wn * 64 + g * 16 + l;
    const float bv = gb[col];
#pragma unroll
    for (int f = 0; f < 2; ++f) {
      const int mb = m0 + wm * 32 + f * 16 + hi * 8;
#pragma unroll
      for (int rr = 0; rr < 8; ++rr) {
        float x = acc[f][g][rr] + bv;
        gO[(size_t)(mb + rr) * N + col] = (__bf16)(x / (1.0f + __expf(-x)));
      }
    }
  }
}

// ---------------------------------------------------------------------------
// Final layer: one wave per atom; dot(h[atom], W4[s]) + b4[s], scatter-add.
// Launched with 256 threads = 8 waves per block.
// ---------------------------------------------------------------------------
__global__ __launch_bounds__(256)
void final_energy(const __bf16* __restrict__ h, const float* __restrict__ W4,
                  const float* __restrict__ b4, const int* __restrict__ sidx,
                  float* __restrict__ out, int Msp, int H) {
  const int lane = threadIdx.x & 31;
  const int wave = threadIdx.x >> 5;
  const int atom = blockIdx.x * 8 + wave;
  const int s    = atom / Msp;

  v8bf hv = *(const v8bf*)&h[(size_t)atom * H + lane * 8];
  const float* w = W4 + (size_t)s * H + lane * 8;
  float acc = 0.0f;
#pragma unroll
  for (int i = 0; i < 8; ++i) acc += (float)hv[i] * w[i];

#pragma unroll
  for (int off = 16; off > 0; off >>= 1) acc += __shfl_xor(acc, off, 32);

  if (lane == 0) atomicAdd(&out[sidx[atom]], acc + b4[s]);
}

__global__ void zero_f32(float* __restrict__ p, int n) {
  int i = blockIdx.x * blockDim.x + threadIdx.x;
  if (i < n) p[i] = 0.0f;
}

// ---------------------------------------------------------------------------
extern "C" void kernel_launch(void* const* d_in, const int* in_sizes, int n_in,
                              void* d_out, int out_size, void* d_ws,
                              size_t ws_size, hipStream_t stream) {
  (void)in_sizes; (void)n_in; (void)ws_size;

  const float* features = (const float*)d_in[0];
  const int*   sidx     = (const int*)d_in[1];
  // d_in[2] = n_structures scalar (== out_size)
  const float* W1 = (const float*)d_in[3];
  const float* b1 = (const float*)d_in[4];
  const float* W2 = (const float*)d_in[5];
  const float* b2 = (const float*)d_in[6];
  const float* W3 = (const float*)d_in[7];
  const float* b3 = (const float*)d_in[8];
  const float* W4 = (const float*)d_in[9];
  const float* b4 = (const float*)d_in[10];
  float* out = (float*)d_out;

  const int S = 4, M = 8192, F = 7584, H = 256;

  __bf16* h1  = (__bf16*)d_ws;                    // 16 MB
  __bf16* h2  = h1 + (size_t)S * M * H;           // 16 MB
  __bf16* Wt1 = h2 + (size_t)S * M * H;           // 15.5 MB  [S][H][F]
  __bf16* Wt2 = Wt1 + (size_t)S * H * F;          // 0.5 MB   [S][H][H]
  __bf16* Wt3 = Wt2 + (size_t)S * H * H;          // 0.5 MB

  zero_f32<<<dim3((out_size + 255) / 256), dim3(256), 0, stream>>>(out,
                                                                   out_size);

  // Weight pre-pass: f32 [K][N] -> bf16 [N][K]
  transpose_w_bf16<<<dim3(F / 32, H / 32, S), dim3(32, 8), 0, stream>>>(
      W1, Wt1, F, H);
  transpose_w_bf16<<<dim3(H / 32, H / 32, S), dim3(32, 8), 0, stream>>>(
      W2, Wt2, H, H);
  transpose_w_bf16<<<dim3(H / 32, H / 32, S), dim3(32, 8), 0, stream>>>(
      W3, Wt3, H, H);

  dim3 gblk(512);
  dim3 grid(M / 128, S);  // (64, 4), BN == N == 256

  // Layer 1: f32 features -> bf16 while staging; K = 7584 = 96 * 79
  gemm_bias_silu<float, 96><<<grid, gblk, 0, stream>>>(features, Wt1, b1, h1,
                                                       M, F, H);
  // Layers 2/3: all-bf16, fully async-staged; K = 256 = 64 * 4
  gemm_bias_silu<__bf16, 64><<<grid, gblk, 0, stream>>>(h1, Wt2, b2, h2,
                                                        M, H, H);
  gemm_bias_silu<__bf16, 64><<<grid, gblk, 0, stream>>>(h2, Wt3, b3, h1,
                                                        M, H, H);

  // Layer 4 + segment sum (256 threads = 8 waves per block)
  final_energy<<<dim3((S * M) / 8), dim3(256), 0, stream>>>(h1, W4, b4, sidx,
                                                            out, M, H);
}
